// FFM_5772436045843
// MI455X (gfx1250) — matrix operand order repaired and checked by reading, not compiled, not run on hardware
//
#include <hip/hip_runtime.h>
#include <hip/hip_bf16.h>

#define B_    16
#define T_    1024
#define IN_   1024
#define MEM_  64
#define OUT_  1024
#define CTX_  64
#define C_    32
#define H_    (2*MEM_ + 2*OUT_)   // 2176
#define BT_   (B_*T_)             // 16384
#define SRK_  (MEM_*CTX_)         // 4096

typedef __attribute__((ext_vector_type(16))) __bf16 v16bf;
typedef __attribute__((ext_vector_type(8)))  float  v8f;

__device__ __forceinline__ unsigned short f2bf(float f) {
  unsigned int u = __float_as_uint(f);
  u += 0x7FFFu + ((u >> 16) & 1u);          // round-to-nearest-even
  return (unsigned short)(u >> 16);
}

__device__ __forceinline__ float sigm(float x) {
  return 1.0f / (1.0f + __expf(-x));
}

__device__ __forceinline__ v16bf load_afrag(const unsigned short* p) {
  // one contiguous 32B chunk (16 consecutive K per lane-half, same order as B)
  struct Q { uint4 a, b; } q;
  q.a = *(const uint4*)(p);
  q.b = *(const uint4*)(p + 8);
  v16bf v; __builtin_memcpy(&v, &q, 32);
  return v;
}

__device__ __forceinline__ v16bf load_bfrag(const unsigned short* p) {
  // one contiguous 32B chunk (16 consecutive K per lane-half, same order as A)
  struct Q { uint4 a, b; } q;
  q.a = *(const uint4*)(p);
  q.b = *(const uint4*)(p + 8);
  v16bf v; __builtin_memcpy(&v, &q, 32);
  return v;
}

// ---------------------------------------------------------------------------
// f32 -> bf16 conversion, 4 elements/thread (n must be divisible by 4)
// ---------------------------------------------------------------------------
__global__ __launch_bounds__(256)
void ffm_cvt_bf16(const float* __restrict__ in, unsigned short* __restrict__ out, int n4) {
  int i = blockIdx.x * blockDim.x + threadIdx.x;
  if (i >= n4) return;
  float4 f = ((const float4*)in)[i];
  uint2 p;
  p.x = (unsigned int)f2bf(f.x) | ((unsigned int)f2bf(f.y) << 16);
  p.y = (unsigned int)f2bf(f.z) | ((unsigned int)f2bf(f.w) << 16);
  ((uint2*)out)[i] = p;
}

// ---------------------------------------------------------------------------
// bf16 WMMA GEMM:  C[M,N] = A[M,K] * B[N,K]^T + bias[N]
// 2x2 register blocking: one wave -> 32x32 output (4 accumulators),
// 8 loads feed 4 WMMAs per K-step. Block = 8 waves (4M x 2N) -> 128x64 tile.
// K % 32 == 0, M % 128 == 0, N % 64 == 0.
// ---------------------------------------------------------------------------
__global__ __launch_bounds__(256)
void ffm_gemm_bf16(const unsigned short* __restrict__ A,
                   const unsigned short* __restrict__ Bm,
                   const float* __restrict__ bias,
                   float* __restrict__ Cout,
                   int M, int N, int K) {
  const int lane = threadIdx.x & 31;
  const int wave = threadIdx.x >> 5;
  const int l16  = lane & 15;
  const int half = lane >> 4;               // 0: lanes 0-15, 1: lanes 16-31

  const int wm = wave & 3;                  // 4 M-subtiles per block
  const int wn = wave >> 2;                 // 2 N-subtiles per block
  const int m0 = blockIdx.y * 128 + wm * 32;
  const int n0 = blockIdx.x * 64  + wn * 32;

  const unsigned short* aRow0 = A  + (size_t)(m0 + l16) * K + half * 16;
  const unsigned short* aRow1 = aRow0 + (size_t)16 * K;
  const unsigned short* bRow0 = Bm + (size_t)(n0 + l16) * K + half * 16;
  const unsigned short* bRow1 = bRow0 + (size_t)16 * K;

  v8f acc00 = {}, acc01 = {}, acc10 = {}, acc11 = {};
  for (int k0 = 0; k0 < K; k0 += 32) {
    __builtin_prefetch(aRow0 + k0 + 128, 0, 0);   // global_prefetch_b8
    __builtin_prefetch(aRow1 + k0 + 128, 0, 0);
    __builtin_prefetch(bRow0 + k0 + 128, 0, 0);
    __builtin_prefetch(bRow1 + k0 + 128, 0, 0);

    v16bf a0 = load_afrag(aRow0 + k0);
    v16bf a1 = load_afrag(aRow1 + k0);
    v16bf b0 = load_bfrag(bRow0 + k0);
    v16bf b1 = load_bfrag(bRow1 + k0);

    acc00 = __builtin_amdgcn_wmma_f32_16x16x32_bf16(false, a0, false, b0,
                                                    (short)0, acc00, false, false);
    acc01 = __builtin_amdgcn_wmma_f32_16x16x32_bf16(false, a0, false, b1,
                                                    (short)0, acc01, false, false);
    acc10 = __builtin_amdgcn_wmma_f32_16x16x32_bf16(false, a1, false, b0,
                                                    (short)0, acc10, false, false);
    acc11 = __builtin_amdgcn_wmma_f32_16x16x32_bf16(false, a1, false, b1,
                                                    (short)0, acc11, false, false);
  }

  // Epilogue: D layout -> VGPR r, lane: row = r + 8*half, col = l16
  const float bb0 = bias[n0 + l16];
  const float bb1 = bias[n0 + 16 + l16];
#pragma unroll
  for (int r = 0; r < 8; ++r) {
    const int mlo = m0 + r + half * 8;
    const int mhi = mlo + 16;
    Cout[(size_t)mlo * N + n0 + l16]      = acc00[r] + bb0;
    Cout[(size_t)mlo * N + n0 + 16 + l16] = acc01[r] + bb1;
    Cout[(size_t)mhi * N + n0 + l16]      = acc10[r] + bb0;
    Cout[(size_t)mhi * N + n0 + 16 + l16] = acc11[r] + bb1;
  }
}

// ---------------------------------------------------------------------------
// Complex linear scan: one wave per (b, m); lane = c.
// Fuses in-gate: y = h[..,m] * sigmoid(h[..,1088+m]).
// Writes states (complex64 interleaved, == reference sr layout) to d_out and
// a bf16 copy of sr for GEMM2.
// ---------------------------------------------------------------------------
__global__ __launch_bounds__(256)
void ffm_scan(const float* __restrict__ h,          // [BT, H]
              const float* __restrict__ a,          // [MEM]
              const float* __restrict__ bfreq,      // [C]
              float* __restrict__ states,           // [BT, MEM, C, 2]
              unsigned short* __restrict__ sr16) {  // [BT, MEM*C*2] bf16
  const int lane = threadIdx.x & 31;                             // c
  const int pair = blockIdx.x * (blockDim.x >> 5) + (threadIdx.x >> 5); // b*MEM+m
  const int b = pair >> 6;
  const int m = pair & 63;

  const float decay = __expf(-fabsf(a[m]));
  const float gr = decay * __cosf(bfreq[lane]);
  const float gi = decay * __sinf(bfreq[lane]);

  float sre = 0.0f, sim = 0.0f;
  for (int t = 0; t < T_; ++t) {
    const size_t bt = (size_t)b * T_ + t;
    const float* hrow = h + bt * H_;
    const float y = hrow[m] * sigm(hrow[MEM_ + OUT_ + m]);

    const float nre = fmaf(gr, sre, fmaf(-gi, sim, y));
    const float nim = fmaf(gr, sim, gi * sre);
    sre = nre; sim = nim;

    ((float2*)states)[(bt * MEM_ + m) * C_ + lane] = make_float2(sre, sim);

    const unsigned int packed =
        (unsigned int)f2bf(sre) | ((unsigned int)f2bf(sim) << 16);
    ((unsigned int*)sr16)[bt * (SRK_ / 2) + m * C_ + lane] = packed;
  }
}

// ---------------------------------------------------------------------------
// LayerNorm + gated mix: one block per (b,t) row of 1024.
// out = LN(z) * out_gate + thru * (1 - out_gate)
// ---------------------------------------------------------------------------
__global__ __launch_bounds__(256)
void ffm_out(const float* __restrict__ z,    // [BT, OUT] (mix_b included)
             const float* __restrict__ h,    // [BT, H]
             float* __restrict__ out) {
  __shared__ float red[2][8];
  const size_t bt = blockIdx.x;
  const float* zr = z + bt * OUT_;
  const float* hr = h + bt * H_;

  float v[4];
  float s = 0.0f, s2 = 0.0f;
#pragma unroll
  for (int i = 0; i < 4; ++i) {
    v[i] = zr[threadIdx.x + i * 256];
    s += v[i];
    s2 += v[i] * v[i];
  }
#pragma unroll
  for (int off = 16; off > 0; off >>= 1) {
    s  += __shfl_xor(s, off, 32);
    s2 += __shfl_xor(s2, off, 32);
  }
  const int wave = threadIdx.x >> 5, lane = threadIdx.x & 31;
  if (lane == 0) { red[0][wave] = s; red[1][wave] = s2; }
  __syncthreads();

  float ts = 0.0f, ts2 = 0.0f;
#pragma unroll
  for (int i = 0; i < 8; ++i) { ts += red[0][i]; ts2 += red[1][i]; }
  const float mu  = ts * (1.0f / OUT_);
  const float var = ts2 * (1.0f / OUT_) - mu * mu;
  const float rs  = rsqrtf(var + 1e-5f);

#pragma unroll
  for (int i = 0; i < 4; ++i) {
    const int o  = threadIdx.x + i * 256;
    const float og   = sigm(hr[MEM_ + OUT_ + MEM_ + o]);   // h[.., 1152+o]
    const float thru = hr[MEM_ + o];                        // h[..,   64+o]
    out[bt * OUT_ + o] = (v[i] - mu) * rs * og + thru * (1.0f - og);
  }
}

// ---------------------------------------------------------------------------
extern "C" void kernel_launch(void* const* d_in, const int* in_sizes, int n_in,
                              void* d_out, int out_size, void* d_ws, size_t ws_size,
                              hipStream_t stream) {
  const float* x     = (const float*)d_in[0];   // [B,T,IN]
  const float* pre_w = (const float*)d_in[1];   // [H, IN]
  const float* pre_b = (const float*)d_in[2];   // [H]
  const float* a     = (const float*)d_in[3];   // [MEM]
  const float* bfrq  = (const float*)d_in[4];   // [C]
  const float* mix_w = (const float*)d_in[5];   // [OUT, SRK]
  const float* mix_b = (const float*)d_in[6];   // [OUT]

  float* out    = (float*)d_out;                         // [BT, OUT]
  float* states = (float*)d_out + (size_t)BT_ * OUT_;    // [BT, MEM, C, 2]

  // Workspace carving (256-byte aligned blocks)
  size_t off = 0;
  auto carve = [&](size_t bytes) -> void* {
    void* p = (char*)d_ws + off;
    off += (bytes + 255) & ~(size_t)255;
    return p;
  };
  unsigned short* xb   = (unsigned short*)carve((size_t)BT_ * IN_  * 2);
  unsigned short* pwb  = (unsigned short*)carve((size_t)H_  * IN_  * 2);
  unsigned short* mwb  = (unsigned short*)carve((size_t)OUT_* SRK_ * 2);
  float*          hbuf = (float*)         carve((size_t)BT_ * H_   * 4);
  unsigned short* sr16 = (unsigned short*)carve((size_t)BT_ * SRK_ * 2);
  float*          zbuf = (float*)         carve((size_t)BT_ * OUT_ * 4);

  // 1) bf16 conversions
  {
    int n4 = BT_ * IN_ / 4;
    ffm_cvt_bf16<<<(n4 + 255) / 256, 256, 0, stream>>>(x, xb, n4);
  }
  {
    int n4 = H_ * IN_ / 4;
    ffm_cvt_bf16<<<(n4 + 255) / 256, 256, 0, stream>>>(pre_w, pwb, n4);
  }
  {
    int n4 = OUT_ * SRK_ / 4;
    ffm_cvt_bf16<<<(n4 + 255) / 256, 256, 0, stream>>>(mix_w, mwb, n4);
  }

  // 2) h = x @ pre_w^T + pre_b     (16384 x 2176, K=1024)
  ffm_gemm_bf16<<<dim3(H_ / 64, BT_ / 128), 256, 0, stream>>>(
      xb, pwb, pre_b, hbuf, BT_, H_, IN_);

  // 3) complex scan -> states (d_out) + sr bf16 (ws)
  ffm_scan<<<(B_ * MEM_) / 8, 256, 0, stream>>>(hbuf, a, bfrq, states, sr16);

  // 4) z = sr @ mix_w^T + mix_b    (16384 x 1024, K=4096)
  ffm_gemm_bf16<<<dim3(OUT_ / 64, BT_ / 128), 256, 0, stream>>>(
      sr16, mwb, mix_b, zbuf, BT_, OUT_, SRK_);

  // 5) LayerNorm + gated output
  ffm_out<<<BT_, 256, 0, stream>>>(zbuf, hbuf, out);
}
